// RecurrentNN_25872882991938
// MI455X (gfx1250) — matrix operand order repaired and checked by reading, not compiled, not run on hardware
//
#include <hip/hip_runtime.h>
#include <hip/hip_bf16.h>
#include <math.h>

#define HIDDEN      4096
#define STEPS       2048
#define NWG         256
#define ROWS_PER_WG (HIDDEN / NWG)   // 16 rows per workgroup
#define TPB         256
#define WAVES       (TPB / 32)       // 8 wave32s
#define CHUNKS_PER_WAVE (HIDDEN / 32 / WAVES)   // 16 K-chunks of 32 per wave

typedef __attribute__((ext_vector_type(16))) __bf16 v16bf;
typedef __attribute__((ext_vector_type(8)))  float  v8f;

// Workspace layout (d_ws). Header padded to 256B so hbuf is 16B aligned.
struct WS {
    unsigned barrier_count;
    unsigned barrier_gen;
    unsigned pad[62];
    __bf16 hbuf[2][HIDDEN];   // double-buffered hidden state, bf16
};

// ---------------------------------------------------------------------------
// Init kernel: runs every launch so graph replays are deterministic.
// Sets d_out[t] = b_out (y accumulators), zeros h0, resets barrier state.
// ---------------------------------------------------------------------------
__global__ void rnn_init(float* out, const float* b_out, WS* ws) {
    int idx = blockIdx.x * blockDim.x + threadIdx.x;
    if (idx == 0) {
        ws->barrier_count = 0u;
        ws->barrier_gen   = 0u;
    }
    if (idx < STEPS)  out[idx] = b_out[0];
    if (idx < HIDDEN) {
        ws->hbuf[0][idx] = (__bf16)0.0f;
        ws->hbuf[1][idx] = (__bf16)0.0f;
    }
}

// ---------------------------------------------------------------------------
// Persistent RNN kernel: one workgroup owns 16 rows of W_hh, kept in LDS as
// bf16. Each step: async h broadcast to LDS -> WMMA mat-vec slice -> tanh ->
// write h slice + partial y, then a device-wide generation barrier.
// ---------------------------------------------------------------------------
__global__ __launch_bounds__(TPB) void rnn_persistent(
        const float* __restrict__ inputs,
        const float* __restrict__ W_ih,
        const float* __restrict__ W_hh,
        const float* __restrict__ b_ih,
        const float* __restrict__ b_hh,
        const float* __restrict__ W_out,
        float* out,     // [STEPS + HIDDEN]
        WS*    ws)
{
    __shared__ __bf16 w_lds[ROWS_PER_WG * HIDDEN];  // 128 KB resident weights
    __shared__ __bf16 h_lds[HIDDEN];                // 8 KB hidden-state copy
    __shared__ float  acc[ROWS_PER_WG];
    __shared__ float  yred[ROWS_PER_WG];

    const int tid  = threadIdx.x;
    const int lane = tid & 31;
    const int row0 = blockIdx.x * ROWS_PER_WG;
    // wave index as a compiler-visible scalar => scalar loop control and
    // immediate ds offsets in the unrolled WMMA loop
    const int wavu = __builtin_amdgcn_readfirstlane(tid >> 5);

    // ---- one-time staging: fp32 W_hh slice -> bf16 LDS (row-major) --------
    for (int i = tid * 4; i < ROWS_PER_WG * HIDDEN; i += TPB * 4) {
        float4 f = *(const float4*)&W_hh[(size_t)row0 * HIDDEN + i];
        union { __bf16 h[4]; uint2 u; } p;
        p.h[0] = (__bf16)f.x; p.h[1] = (__bf16)f.y;
        p.h[2] = (__bf16)f.z; p.h[3] = (__bf16)f.w;
        *(uint2*)&w_lds[i] = p.u;    // single ds_store_b64
    }
    __syncthreads();

    // per-lane WMMA A/B addressing constants (16-bit operand layouts)
    const int arow   = lane & 15;
    const int koff_a = (lane < 16) ? 0 : 8;    // A: VGPR0-3 hold K 0..7 / 8..15
    const int koff_b = (lane < 16) ? 0 : 8;    // B: same K split as A (0..7,16..23 / 8..15,24..31)
    const __bf16* abase = &w_lds[arow * HIDDEN + wavu * 32 + koff_a];
    const __bf16* bbase = &h_lds[wavu * 32 + koff_b];

    // LDS byte offset of h_lds (low 32 bits of the generic pointer) and the
    // per-thread byte offset for the async global->LDS broadcast.
    const unsigned h_lds_off = (unsigned)(uintptr_t)(const void*)&h_lds[0];
    const unsigned tb_off    = (unsigned)tid * 16u;  // 2 x b128 per thread

    for (int t = 0; t < STEPS; ++t) {
        // ---- 1. async broadcast of current h (bf16, 8 KB) into LDS --------
        {
            const __bf16* hsrc = ws->hbuf[t & 1];
            // INST_OFFSET is added to both the LDS and global address, so one
            // VGPR pair covers both halves of the 8 KB tile.
            asm volatile(
                "global_load_async_to_lds_b128 %0, %1, %2\n\t"
                "global_load_async_to_lds_b128 %0, %1, %2 offset:4096"
                :
                : "v"(h_lds_off + tb_off), "v"(tb_off), "s"(hsrc)
                : "memory");
        }
        if (tid < ROWS_PER_WG) acc[tid] = 0.0f;
        asm volatile("s_wait_asynccnt 0" ::: "memory");
        __syncthreads();

        // ---- 2. WMMA mat-vec: 16 rows x 4096 dot, split over 8 waves ------
        v8f d = (v8f){0.f, 0.f, 0.f, 0.f, 0.f, 0.f, 0.f, 0.f};
#pragma unroll
        for (int j = 0; j < CHUNKS_PER_WAVE; ++j) {
            union { v16bf v; uint4 q[2]; } a, b;
            // A tile (16x32 bf16): row = arow, K split per lane half
            a.q[0] = *(const uint4*)(abase + j * (WAVES * 32));        // K lo
            a.q[1] = *(const uint4*)(abase + j * (WAVES * 32) + 16);   // K hi
            // B tile (32x16 bf16): h chunk replicated across all 16 columns
            b.q[0] = *(const uint4*)(bbase + j * (WAVES * 32));        // K lo
            b.q[1] = *(const uint4*)(bbase + j * (WAVES * 32) + 16);   // K hi
            d = __builtin_amdgcn_wmma_f32_16x16x32_bf16(
                    false, a.v, false, b.v, (short)0, d, false, false);
        }

        // ---- 3. reduce wave partials: lane 0 -> rows 0..7, lane 16 -> 8..15
        if (lane == 0 || lane == 16) {
            const int mbase = (lane == 16) ? 8 : 0;
#pragma unroll
            for (int v = 0; v < 8; ++v)
                atomicAdd(&acc[mbase + v], d[v]);
        }
        __syncthreads();

        // ---- 4. finish rows: bias + x*W_ih, tanh, emit h + partial y ------
        if (tid < ROWS_PER_WG) {
            const float x = (t == 0) ? inputs[0] : out[t - 1];
            const int row = row0 + tid;
            float hpre = acc[tid] + b_ih[row] + b_hh[row] + x * W_ih[row];
            float hnew = tanhf(hpre);
            ws->hbuf[(t + 1) & 1][row] = (__bf16)hnew;
            yred[tid] = W_out[row] * hnew;
            if (t == STEPS - 1) out[STEPS + row] = hnew;  // h_final (fp32)
        }
        __threadfence();   // release this thread's global stores
        __syncthreads();

        // ---- 5. partial readout + device-wide generation barrier ----------
        if (tid == 0) {
            float s = 0.0f;
#pragma unroll
            for (int r = 0; r < ROWS_PER_WG; ++r) s += yred[r];
            atomicAdd(&out[t], s);   // out[t] preloaded with b_out
            __threadfence();

            const unsigned target = (unsigned)(t + 1);
            unsigned ticket = __hip_atomic_fetch_add(
                &ws->barrier_count, 1u, __ATOMIC_ACQ_REL, __HIP_MEMORY_SCOPE_AGENT);
            if (ticket == NWG - 1) {
                __hip_atomic_store(&ws->barrier_count, 0u,
                                   __ATOMIC_RELAXED, __HIP_MEMORY_SCOPE_AGENT);
                __hip_atomic_fetch_add(&ws->barrier_gen, 1u,
                                       __ATOMIC_ACQ_REL, __HIP_MEMORY_SCOPE_AGENT);
            } else {
                while (__hip_atomic_load(&ws->barrier_gen, __ATOMIC_ACQUIRE,
                                         __HIP_MEMORY_SCOPE_AGENT) < target) {
                    __builtin_amdgcn_s_sleep(2);
                }
            }
        }
        __syncthreads();
    }
}

// ---------------------------------------------------------------------------
extern "C" void kernel_launch(void* const* d_in, const int* in_sizes, int n_in,
                              void* d_out, int out_size, void* d_ws, size_t ws_size,
                              hipStream_t stream) {
    const float* inputs = (const float*)d_in[0];
    const float* W_ih   = (const float*)d_in[1];
    const float* W_hh   = (const float*)d_in[2];
    const float* b_ih   = (const float*)d_in[3];
    const float* b_hh   = (const float*)d_in[4];
    const float* W_out  = (const float*)d_in[5];
    const float* b_out  = (const float*)d_in[6];
    float* out = (float*)d_out;
    WS* ws = (WS*)d_ws;

    rnn_init<<<16, 256, 0, stream>>>(out, b_out, ws);
    rnn_persistent<<<NWG, TPB, 0, stream>>>(inputs, W_ih, W_hh, b_ih, b_hh,
                                            W_out, out, ws);
}
